// COHF_67001489817741
// MI455X (gfx1250) — matrix operand order repaired and checked by reading, not compiled, not run on hardware
//
#include <hip/hip_runtime.h>
#include <hip/hip_bf16.h>

typedef __attribute__((ext_vector_type(16))) __bf16 v16bf;
typedef __attribute__((ext_vector_type(8)))  __bf16 v8bf;
typedef __attribute__((ext_vector_type(8)))  float  v8f;

#define N_NODES 100000
#define NFEAT   256
#define E1DIM   128
#define E2DIM   64
#define NEDGE   500000

// ---------------------------------------------------------------------------
// Kernel 0: pack W1 (f32, row-major [256,128]) into bf16 in this GEMM's contiguous-K WMMA
// B-operand register layout for V_WMMA_F32_16X16X32_BF16:
//   lane L: N = (L&15) + 16*tile_n ; K = e + 16*(L>>4) + 32*chunk  (e = 0..15)
// Flattened as [tile_n][chunk][lane][e] so the GEMM loads 32B/lane contiguous.
// ---------------------------------------------------------------------------
__global__ void pack_w1_kernel(const float* __restrict__ W1,
                               __bf16* __restrict__ W1p) {
    int g      = blockIdx.x * blockDim.x + threadIdx.x;  // 0 .. 32767
    int e      = g & 15;
    int lane   = (g >> 4) & 31;
    int c      = (g >> 9) & 7;
    int tile_n = g >> 12;
    int k = 32 * c + 16 * (lane >> 4) + e;
    int n = 16 * tile_n + (lane & 15);
    W1p[g] = (__bf16)W1[k * E1DIM + n];
}

// ---------------------------------------------------------------------------
// Kernel 1: fused node pass.  Per block: 16 nodes, 256 threads = 8 waves.
//   e1 = relu(feats @ W1 + b1)       (bf16 WMMA, f32 accumulate, never stored)
//   A[n] = e1[n]·W2[0:128]   + e2c[n]·W2[256:320] + e2d[n]·W2[320:384]
//   B[n] = e1[n]·W2[128:256] + e2c[n]·W2[384:448] + e2d[n]·W2[448:512]
// ---------------------------------------------------------------------------
__global__ __launch_bounds__(256)
void node_kernel(const float* __restrict__ feats,
                 const float* __restrict__ e2c,
                 const float* __restrict__ e2d,
                 const __bf16* __restrict__ W1p,
                 const float* __restrict__ b1,
                 const float* __restrict__ W2,
                 float* __restrict__ Aout,
                 float* __restrict__ Bout) {
    __shared__ __bf16 s_a[16][264];   // 16 rows x 256 cols bf16, +8 pad (bank-conflict free)
    __shared__ float  s_pa[16];
    __shared__ float  s_pb[16];

    const int t    = threadIdx.x;
    const int base = blockIdx.x * 16;   // N_NODES = 6250 * 16 exactly

    // --- cooperative stage of the 16x256 feats tile into LDS as bf16 ---
    {
        const int row = t >> 4;
        const int c0  = (t & 15) << 4;
        const float4* src = (const float4*)(feats + (size_t)(base + row) * NFEAT + c0);
        __bf16* dst = &s_a[row][c0];
        #pragma unroll
        for (int q = 0; q < 4; ++q) {
            float4 v = src[q];
            dst[4 * q + 0] = (__bf16)v.x;
            dst[4 * q + 1] = (__bf16)v.y;
            dst[4 * q + 2] = (__bf16)v.z;
            dst[4 * q + 3] = (__bf16)v.w;
        }
    }

    // --- e2 contribution: seed s_pa / s_pb (16 threads per node) ---
    {
        const int row = t >> 4;
        const int k   = t & 15;
        const float* pc = e2c + (size_t)(base + row) * E2DIM;
        const float* pd = e2d + (size_t)(base + row) * E2DIM;
        float pa = 0.f, pb = 0.f;
        #pragma unroll
        for (int s = 0; s < 4; ++s) {
            int j = k + 16 * s;
            float cv = pc[j], dv = pd[j];
            pa += cv * W2[256 + j] + dv * W2[320 + j];
            pb += cv * W2[384 + j] + dv * W2[448 + j];
        }
        #pragma unroll
        for (int off = 8; off >= 1; off >>= 1) {
            pa += __shfl_xor(pa, off, 16);
            pb += __shfl_xor(pb, off, 16);
        }
        if (k == 0) { s_pa[row] = pa; s_pb[row] = pb; }
    }
    __syncthreads();

    // --- GEMM: each wave owns one 16-wide column tile of e1, K loop = 8x32 ---
    const int wave = t >> 5;      // 0..7 : column tile
    const int lane = t & 31;
    const int h    = lane >> 4;   // lane half (selects K sub-ranges)
    const int m    = lane & 15;   // A-operand row / C-operand column index

    v8f acc = {0.f, 0.f, 0.f, 0.f, 0.f, 0.f, 0.f, 0.f};
    #pragma unroll
    for (int c = 0; c < 8; ++c) {
        // A fragment: contiguous K = 32c+16h+{0..15} (matches packed B)
        v8bf lo = *(const v8bf*)&s_a[m][32 * c + 16 * h];
        v8bf hi = *(const v8bf*)&s_a[m][32 * c + 16 * h + 8];
        v16bf a;
        #pragma unroll
        for (int i = 0; i < 8; ++i) { a[i] = lo[i]; a[8 + i] = hi[i]; }
        // B fragment: pre-packed, 32 contiguous bytes per lane
        v16bf b = *(const v16bf*)(W1p + (((wave * 8 + c) * 32 + lane) << 4));
        acc = __builtin_amdgcn_wmma_f32_16x16x32_bf16(
            false, a, false, b, (short)0, acc, false, false);
    }

    // --- epilogue: relu(acc + b1) dotted with W2[0:128] / W2[128:256] ---
    const int col = wave * 16 + m;      // C/D layout: lane holds column (lane&15)
    const float b1v = b1[col];
    const float w2a = W2[col];
    const float w2b = W2[128 + col];
    #pragma unroll
    for (int r = 0; r < 8; ++r) {       // C/D layout: row = r + 8*(lane>>4)
        float v = acc[r] + b1v;
        v = v > 0.f ? v : 0.f;
        float pa = v * w2a;
        float pb = v * w2b;
        #pragma unroll
        for (int off = 8; off >= 1; off >>= 1) {
            pa += __shfl_xor(pa, off, 16);
            pb += __shfl_xor(pb, off, 16);
        }
        if (m == 0) {
            int rowi = r + 8 * h;
            atomicAdd(&s_pa[rowi], pa);
            atomicAdd(&s_pb[rowi], pb);
        }
    }
    __syncthreads();

    if (t < 16) {
        Aout[base + t] = s_pa[t];
        Bout[base + t] = s_pb[t];
    }
}

// ---------------------------------------------------------------------------
// Kernel 2: edge pass.  out[e] = A[i] + B[j] + b2   (A/B are L2-resident)
// edge_index is [2, E] row-major int32: row 0 = i, row 1 = j.
// ---------------------------------------------------------------------------
__global__ void edge_kernel(const int* __restrict__ idx,
                            const float* __restrict__ A,
                            const float* __restrict__ B,
                            const float* __restrict__ b2,
                            float* __restrict__ out, int E) {
    int tid = blockIdx.x * blockDim.x + threadIdx.x;
    if (tid < E) {
        out[tid] = A[idx[tid]] + B[idx[tid + E]] + b2[0];
    }
}

extern "C" void kernel_launch(void* const* d_in, const int* in_sizes, int n_in,
                              void* d_out, int out_size, void* d_ws, size_t ws_size,
                              hipStream_t stream) {
    const float* feats = (const float*)d_in[0];
    const float* e2c   = (const float*)d_in[1];
    const float* e2d   = (const float*)d_in[2];
    const float* W1    = (const float*)d_in[3];
    const float* b1    = (const float*)d_in[4];
    const float* W2    = (const float*)d_in[5];
    const float* b2    = (const float*)d_in[6];
    const int*   eip   = (const int*)d_in[7];
    const int*   ein   = (const int*)d_in[8];
    float* out = (float*)d_out;

    // workspace layout: [0,64K) packed W1 bf16 ; then A[100000] f32 ; B[100000] f32
    __bf16* W1p = (__bf16*)d_ws;
    float*  Aout = (float*)((char*)d_ws + 65536);
    float*  Bout = Aout + N_NODES;

    pack_w1_kernel<<<128, 256, 0, stream>>>(W1, W1p);
    node_kernel<<<N_NODES / 16, 256, 0, stream>>>(feats, e2c, e2d, W1p, b1, W2,
                                                  Aout, Bout);
    edge_kernel<<<(NEDGE + 255) / 256, 256, 0, stream>>>(eip, Aout, Bout, b2,
                                                         out, NEDGE);
    edge_kernel<<<(NEDGE + 255) / 256, 256, 0, stream>>>(ein, Aout, Bout, b2,
                                                         out + NEDGE, NEDGE);
}
